// PropertyPredictor_29566554866306
// MI455X (gfx1250) — matrix operand order repaired and checked by reading, not compiled, not run on hardware
//
#include <hip/hip_runtime.h>

typedef __attribute__((ext_vector_type(16))) _Float16 v16h;
typedef __attribute__((ext_vector_type(8)))  _Float16 v8h;
typedef __attribute__((ext_vector_type(8)))  float    v8f;

#define H 32
#define NG 2000

__device__ __forceinline__ float sigm(float x) { return 1.0f / (1.0f + __expf(-x)); }

// ---------------- node encoder: h = nf @ enc_W + enc_b ----------------
__global__ void enc_kernel(const float* __restrict__ nf, const float* __restrict__ W,
                           const float* __restrict__ b, float* __restrict__ h, int N) {
    int t = blockIdx.x * blockDim.x + threadIdx.x;
    int n = t >> 5, i = t & 31;
    if (n >= N) return;
    float acc = b[i];
    #pragma unroll
    for (int k = 0; k < 16; ++k) acc += nf[(size_t)n * 16 + k] * W[k * 32 + i];
    h[(size_t)n * 32 + i] = acc;
}

// ---------------- edge MLP layer 1: R = relu(ef @ W1 + b1) -> f16 ----------------
__global__ void edgefeat_kernel(const float* __restrict__ ef, const float* __restrict__ W1,
                                const float* __restrict__ b1, _Float16* __restrict__ R, int E) {
    int t = blockIdx.x * blockDim.x + threadIdx.x;
    int e = t >> 6, k = t & 63;
    if (e >= E) return;
    float acc = b1[k];
    #pragma unroll
    for (int d = 0; d < 8; ++d) acc += ef[(size_t)e * 8 + d] * W1[d * 64 + k];
    R[(size_t)e * 64 + k] = (_Float16)fmaxf(acc, 0.0f);
}

// ---------------- pack e_W2 into per-lane WMMA B-fragment layout ----------------
// layout index = ((ntile*2 + kstep)*32 + lane)*16 + t ; element = W2[k][col]
// with k = kstep*32 + (lane>=16 ? 16 : 0) + t ; col = ntile*16 + (lane&15)
__global__ void packw2_kernel(const float* __restrict__ W2, _Float16* __restrict__ W2p) {
    int idx = blockIdx.x * blockDim.x + threadIdx.x;
    if (idx >= 65536) return;
    int t    = idx & 15;
    int lane = (idx >> 4) & 31;
    int ks   = (idx >> 9) & 1;
    int nt   = idx >> 10;
    int k    = ks * 32 + ((lane >= 16) ? 16 : 0) + t;
    int col  = nt * 16 + (lane & 15);
    W2p[idx] = (_Float16)W2[k * 1024 + col];
}

// ---------------- fused edge-net GEMM (WMMA) + per-edge matvec + scatter ----------------
// One wave per 16-edge tile. A_tile[16 x 1024] = R_tile[16x64] @ W2[64x1024] + b2,
// computed in two column halves (2 x 32 n-tiles) with v_wmma_f32_16x16x32_f16 so the
// LDS footprint is 16x520 f16 (~19 KB/workgroup -> ~16 waves resident per WGP).
// Rows i<16 of the per-edge 32x32 matrix only use cols <512; rows i>=16 only cols >=512,
// so each half is consumed by a matvec over rows [16h,16h+16) before being overwritten.
__global__ __launch_bounds__(32)
void msg_kernel(const _Float16* __restrict__ R, const _Float16* __restrict__ W2p,
                const float* __restrict__ b2, const float* __restrict__ h,
                const int* __restrict__ ei, float* __restrict__ mbuf, int E) {
    __shared__ _Float16 Alds[16 * 520];   // row = edge-in-tile, col = local (i*32+j) ; pad 8 halves
    __shared__ float    hlds[16 * 33];    // gathered h[src], padded stride
    __shared__ int      sidx[16], tidx[16];

    const int lane = threadIdx.x;
    const int base = blockIdx.x * 16;

    if (lane < 16) {
        int e = base + lane; if (e >= E) e = E - 1;
        sidx[lane] = ei[e];
        tidx[lane] = ei[(size_t)E + e];
    }
    __syncthreads();

    // gather source-node hidden rows (coalesced, one row per iteration)
    for (int k = 0; k < 16; ++k)
        hlds[k * 33 + lane] = h[(size_t)sidx[k] * 32 + lane];

    // A-operand fragments (16x32 f16 layout): 16 contiguous halves per lane, k = 16*(lane/16) + s
    const int m = lane & 15;
    int erow = base + m; if (erow >= E) erow = E - 1;
    const int kb = (lane < 16) ? 0 : 16;
    const _Float16* rrow = R + (size_t)erow * 64;
    v8h a0lo = *(const v8h*)(rrow + kb);
    v8h a0hi = *(const v8h*)(rrow + kb + 8);
    v8h a1lo = *(const v8h*)(rrow + kb + 32);
    v8h a1hi = *(const v8h*)(rrow + kb + 40);
    v16h a0 = __builtin_shufflevector(a0lo, a0hi, 0,1,2,3,4,5,6,7,8,9,10,11,12,13,14,15);
    v16h a1 = __builtin_shufflevector(a1lo, a1hi, 0,1,2,3,4,5,6,7,8,9,10,11,12,13,14,15);

    const int rbase = (lane < 16) ? 0 : 8;
    const int rgrp  = lane >> 4;          // matvec row-group within each half
    const bool valid = (base + m) < E;
    const int tg = tidx[m];

    for (int half = 0; half < 2; ++half) {
        // ---- GEMM: 32 n-tiles of this column half ----
        for (int ntl = 0; ntl < 32; ++ntl) {
            int nt = half * 32 + ntl;
            v16h b0 = *(const v16h*)(W2p + ((size_t)(nt * 2 + 0) * 32 + lane) * 16);
            v16h b1 = *(const v16h*)(W2p + ((size_t)(nt * 2 + 1) * 32 + lane) * 16);
            float bias = b2[nt * 16 + (lane & 15)];
            v8f c = {bias, bias, bias, bias, bias, bias, bias, bias};
            c = __builtin_amdgcn_wmma_f32_16x16x32_f16(false, a0, false, b0, (short)0, c, false, false);
            c = __builtin_amdgcn_wmma_f32_16x16x32_f16(false, a1, false, b1, (short)0, c, false, false);
            int colL = ntl * 16 + (lane & 15);
            #pragma unroll
            for (int r = 0; r < 8; ++r)
                Alds[(rbase + r) * 520 + colL] = (_Float16)c[r];
        }
        __syncthreads();

        // ---- per-edge matvec for output rows [half*16, half*16+16) ----
        // lane -> (edge m = lane&15, rows rgrp*8 .. rgrp*8+7 of this half)
        #pragma unroll
        for (int i = 0; i < 8; ++i) {
            int rowL = rgrp * 8 + i;                 // local row in [0,16)
            int row  = half * 16 + rowL;             // output channel
            const _Float16* ap = Alds + m * 520 + rowL * 32;
            const float*    hp = hlds + m * 33;
            float acc = 0.0f;
            #pragma unroll
            for (int j = 0; j < 32; ++j) acc += (float)ap[j] * hp[j];
            if (valid) atomicAdd(&mbuf[(size_t)tg * 32 + row], acc);
        }
        __syncthreads();
    }
}

// ---------------- GRU cell: one wave per node, lane = channel ----------------
__global__ __launch_bounds__(256)
void gru_kernel(const float* __restrict__ m, float* __restrict__ h,
                const float* __restrict__ Wih, const float* __restrict__ Whh,
                const float* __restrict__ bih, const float* __restrict__ bhh, int N) {
    __shared__ float sWih[96 * 33], sWhh[96 * 33], sbih[96], sbhh[96];
    for (int idx = threadIdx.x; idx < 96 * 32; idx += 256) {
        int r = idx >> 5, c = idx & 31;
        sWih[r * 33 + c] = Wih[idx];
        sWhh[r * 33 + c] = Whh[idx];
    }
    for (int idx = threadIdx.x; idx < 96; idx += 256) { sbih[idx] = bih[idx]; sbhh[idx] = bhh[idx]; }
    __syncthreads();

    int wave = threadIdx.x >> 5, lane = threadIdx.x & 31;
    int node = blockIdx.x * 8 + wave;
    if (node >= N) return;
    float mv = m[(size_t)node * 32 + lane];
    float hv = h[(size_t)node * 32 + lane];
    float ir = sbih[lane], iz = sbih[32 + lane], inn = sbih[64 + lane];
    float hr = sbhh[lane], hz = sbhh[32 + lane], hn  = sbhh[64 + lane];
    #pragma unroll 8
    for (int j = 0; j < 32; ++j) {
        float mj = __shfl(mv, j, 32);
        float hj = __shfl(hv, j, 32);
        ir  += mj * sWih[lane * 33 + j];
        iz  += mj * sWih[(32 + lane) * 33 + j];
        inn += mj * sWih[(64 + lane) * 33 + j];
        hr  += hj * sWhh[lane * 33 + j];
        hz  += hj * sWhh[(32 + lane) * 33 + j];
        hn  += hj * sWhh[(64 + lane) * 33 + j];
    }
    float r = sigm(ir + hr);
    float z = sigm(iz + hz);
    float n = tanhf(inn + r * hn);
    h[(size_t)node * 32 + lane] = (1.0f - z) * n + z * hv;
}

// ---------------- Set2Set pieces ----------------
__device__ __forceinline__ unsigned fmap(float x) {
    unsigned b = __float_as_uint(x);
    return b ^ ((b & 0x80000000u) ? 0xFFFFFFFFu : 0x80000000u);
}
__device__ __forceinline__ float funmap(unsigned k) {
    unsigned b = (k & 0x80000000u) ? (k ^ 0x80000000u) : ~k;
    return __uint_as_float(b);
}

__global__ void s2s_dot_kernel(const float* __restrict__ h, const float* __restrict__ sh,
                               const int* __restrict__ batch, float* __restrict__ e,
                               unsigned* __restrict__ gmax, int N) {
    int n = blockIdx.x * blockDim.x + threadIdx.x;
    if (n >= N) return;
    int b = batch[n];
    float acc = 0.0f;
    #pragma unroll 8
    for (int j = 0; j < 32; ++j) acc += h[(size_t)n * 32 + j] * sh[(size_t)b * 32 + j];
    e[n] = acc;
    atomicMax(&gmax[b], fmap(acc));
}

__global__ void s2s_exp_kernel(float* __restrict__ e, const unsigned* __restrict__ gmax,
                               const int* __restrict__ batch, float* __restrict__ denom, int N) {
    int n = blockIdx.x * blockDim.x + threadIdx.x;
    if (n >= N) return;
    int b = batch[n];
    float ex = __expf(e[n] - funmap(gmax[b]));
    e[n] = ex;
    atomicAdd(&denom[b], ex);
}

__global__ void s2s_scatter_kernel(const float* __restrict__ ex, const float* __restrict__ denom,
                                   const int* __restrict__ batch, const float* __restrict__ h,
                                   float* __restrict__ rread, int N) {
    int t = blockIdx.x * blockDim.x + threadIdx.x;
    int n = t >> 5, i = t & 31;
    if (n >= N) return;
    int b = batch[n];
    float a = ex[n] / denom[b];
    atomicAdd(&rread[(size_t)b * 32 + i], a * h[(size_t)n * 32 + i]);
}

__global__ __launch_bounds__(256)
void s2s_lstm_kernel(const float* __restrict__ shOld, const float* __restrict__ rread,
                     float* __restrict__ cst, float* __restrict__ shNew,
                     const float* __restrict__ Wih, const float* __restrict__ Whh,
                     const float* __restrict__ bih, const float* __restrict__ bhh, int G) {
    __shared__ float sWih[128 * 65], sWhh[128 * 33];
    for (int idx = threadIdx.x; idx < 128 * 64; idx += 256)
        sWih[(idx >> 6) * 65 + (idx & 63)] = Wih[idx];
    for (int idx = threadIdx.x; idx < 128 * 32; idx += 256)
        sWhh[(idx >> 5) * 33 + (idx & 31)] = Whh[idx];
    __syncthreads();

    int t = blockIdx.x * 256 + threadIdx.x;
    int g = t >> 5, i = t & 31;
    if (g >= G) return;
    const float* x0 = shOld + (size_t)g * 32;
    const float* x1 = rread + (size_t)g * 32;
    float gate[4];
    #pragma unroll
    for (int q = 0; q < 4; ++q) {
        int row = q * 32 + i;
        float acc = bih[row] + bhh[row];
        #pragma unroll 8
        for (int j = 0; j < 32; ++j) {
            acc += x0[j] * sWih[row * 65 + j];
            acc += x1[j] * sWih[row * 65 + 32 + j];
            acc += x0[j] * sWhh[row * 33 + j];
        }
        gate[q] = acc;
    }
    float ig = sigm(gate[0]), fg = sigm(gate[1]);
    float gg = tanhf(gate[2]), og = sigm(gate[3]);
    float cn = fg * cst[(size_t)g * 32 + i] + ig * gg;
    cst[(size_t)g * 32 + i] = cn;
    shNew[(size_t)g * 32 + i] = og * tanhf(cn);
}

// ---------------- output MLP ----------------
__global__ void out_kernel(const float* __restrict__ sh, const float* __restrict__ rread,
                           const float* __restrict__ W1, const float* __restrict__ b1,
                           const float* __restrict__ W2, const float* __restrict__ b2,
                           float* __restrict__ out, int G) {
    int g = blockIdx.x * blockDim.x + threadIdx.x;
    if (g >= G) return;
    float hid[32];
    #pragma unroll
    for (int i = 0; i < 32; ++i) {
        float acc = b1[i];
        for (int j = 0; j < 32; ++j) {
            acc += sh[(size_t)g * 32 + j] * W1[j * 32 + i];
            acc += rread[(size_t)g * 32 + j] * W1[(32 + j) * 32 + i];
        }
        hid[i] = fmaxf(acc, 0.0f);
    }
    #pragma unroll
    for (int o = 0; o < 3; ++o) {
        float acc = b2[o];
        for (int i = 0; i < 32; ++i) acc += hid[i] * W2[i * 3 + o];
        out[(size_t)g * 3 + o] = acc;
    }
}

extern "C" void kernel_launch(void* const* d_in, const int* in_sizes, int n_in,
                              void* d_out, int out_size, void* d_ws, size_t ws_size,
                              hipStream_t stream) {
    const float* nf    = (const float*)d_in[0];
    const float* ef    = (const float*)d_in[1];
    const float* encW  = (const float*)d_in[2];
    const float* encB  = (const float*)d_in[3];
    const float* eW1   = (const float*)d_in[4];
    const float* eB1   = (const float*)d_in[5];
    const float* eW2   = (const float*)d_in[6];
    const float* eB2   = (const float*)d_in[7];
    const float* gWih  = (const float*)d_in[8];
    const float* gWhh  = (const float*)d_in[9];
    const float* gBih  = (const float*)d_in[10];
    const float* gBhh  = (const float*)d_in[11];
    const float* lWih  = (const float*)d_in[12];
    const float* lWhh  = (const float*)d_in[13];
    const float* lBih  = (const float*)d_in[14];
    const float* lBhh  = (const float*)d_in[15];
    const float* oW1   = (const float*)d_in[16];
    const float* oB1   = (const float*)d_in[17];
    const float* oW2   = (const float*)d_in[18];
    const float* oB2   = (const float*)d_in[19];
    const int*   eidx  = (const int*)d_in[20];
    const int*   batch = (const int*)d_in[21];

    const int N = in_sizes[0] / 16;
    const int E = in_sizes[1] / 8;
    const int G = NG;

    // workspace carve (total ~53 MB)
    char* w = (char*)d_ws;
    auto carve = [&](size_t bytes) { void* p = (void*)w; w += (bytes + 255) & ~(size_t)255; return p; };
    _Float16* R    = (_Float16*)carve((size_t)E * 64 * 2);
    _Float16* W2p  = (_Float16*)carve((size_t)65536 * 2);
    float*    h    = (float*)carve((size_t)N * 32 * 4);
    float*    mb   = (float*)carve((size_t)N * 32 * 4);
    float*    ebuf = (float*)carve((size_t)N * 4);
    unsigned* gmax = (unsigned*)carve((size_t)G * 4);
    float*    den  = (float*)carve((size_t)G * 4);
    float*    rrd  = (float*)carve((size_t)G * 32 * 4);
    float*    shA  = (float*)carve((size_t)G * 32 * 4);
    float*    shB  = (float*)carve((size_t)G * 32 * 4);
    float*    sc   = (float*)carve((size_t)G * 32 * 4);
    (void)ws_size; (void)n_in; (void)out_size;

    enc_kernel<<<((size_t)N * 32 + 255) / 256, 256, 0, stream>>>(nf, encW, encB, h, N);
    edgefeat_kernel<<<((size_t)E * 64 + 255) / 256, 256, 0, stream>>>(ef, eW1, eB1, R, E);
    packw2_kernel<<<256, 256, 0, stream>>>(eW2, W2p);

    for (int it = 0; it < 3; ++it) {
        hipMemsetAsync(mb, 0, (size_t)N * 32 * 4, stream);
        msg_kernel<<<(E + 15) / 16, 32, 0, stream>>>(R, W2p, eB2, h, eidx, mb, E);
        gru_kernel<<<(N + 7) / 8, 256, 0, stream>>>(mb, h, gWih, gWhh, gBih, gBhh, N);
    }

    hipMemsetAsync(shA, 0, (size_t)G * 32 * 4, stream);
    hipMemsetAsync(sc,  0, (size_t)G * 32 * 4, stream);
    float* hr = shA;
    float* hw = shB;
    for (int s = 0; s < 4; ++s) {
        hipMemsetAsync(gmax, 0, (size_t)G * 4, stream);
        hipMemsetAsync(den,  0, (size_t)G * 4, stream);
        hipMemsetAsync(rrd,  0, (size_t)G * 32 * 4, stream);
        s2s_dot_kernel<<<(N + 255) / 256, 256, 0, stream>>>(h, hr, batch, ebuf, gmax, N);
        s2s_exp_kernel<<<(N + 255) / 256, 256, 0, stream>>>(ebuf, gmax, batch, den, N);
        s2s_scatter_kernel<<<((size_t)N * 32 + 255) / 256, 256, 0, stream>>>(ebuf, den, batch, h, rrd, N);
        s2s_lstm_kernel<<<((size_t)G * 32 + 255) / 256, 256, 0, stream>>>(hr, rrd, sc, hw, lWih, lWhh, lBih, lBhh, G);
        float* tmp = hr; hr = hw; hw = tmp;
    }

    out_kernel<<<(G + 255) / 256, 256, 0, stream>>>(hr, rrd, oW1, oB1, oW2, oB2, (float*)d_out, G);
}
